// QLinear_68710886802125
// MI455X (gfx1250) — 8-bit matrix operand order repaired and checked by reading, compiled, not run on hardware
//
#include <hip/hip_runtime.h>
#include <stdint.h>

#define IN_CH   4096
#define OUT_CH  4096
#define KW      (IN_CH / 4)
#define MAXQ    255.0f

typedef __attribute__((ext_vector_type(8))) int v8i;
typedef __attribute__((ext_vector_type(4))) int v4i;
typedef __attribute__((ext_vector_type(2))) int v2i;

typedef const uint8_t __attribute__((address_space(1))) gu8;
typedef const v2i     __attribute__((address_space(1))) gv2i;
typedef const v4i     __attribute__((address_space(1))) gv4i;

__global__ __launch_bounds__(256)
void quantize_kernel(const float* __restrict__ x,
                     uint8_t* __restrict__ xq,
                     float* __restrict__ xscale,
                     float* __restrict__ rowsum)
{
    const int row  = blockIdx.x;
    const int tid  = threadIdx.x;
    const int wave = tid >> 5;
    const int lane = tid & 31;
    const float* xr = x + (size_t)row * IN_CH;

    float4 v[4];
    const float4* p = reinterpret_cast<const float4*>(xr) + tid * 4;
#pragma unroll
    for (int i = 0; i < 4; ++i) v[i] = p[i];

    float mx = -3.0e38f;
#pragma unroll
    for (int i = 0; i < 4; ++i) {
        mx = fmaxf(mx, fmaxf(fmaxf(v[i].x, v[i].y), fmaxf(v[i].z, v[i].w)));
    }

    __shared__ float redf[8];
    __shared__ int   redi[8];

#pragma unroll
    for (int off = 16; off > 0; off >>= 1) {
        mx = fmaxf(mx, __shfl_xor(mx, off, 32));
    }
    if (lane == 0) redf[wave] = mx;
    __syncthreads();
    float xmax = redf[0];
#pragma unroll
    for (int i = 1; i < 8; ++i) xmax = fmaxf(xmax, redf[i]);

    const float denom = fmaxf(xmax, 1e-8f);
    const float scale = denom / MAXQ;
    const float inv   = MAXQ / denom;

    int lsum = 0;
    uint32_t w[4];
#pragma unroll
    for (int i = 0; i < 4; ++i) {
        int q0 = (int)fminf(fmaxf(__builtin_rintf(v[i].x * inv), 0.0f), MAXQ);
        int q1 = (int)fminf(fmaxf(__builtin_rintf(v[i].y * inv), 0.0f), MAXQ);
        int q2 = (int)fminf(fmaxf(__builtin_rintf(v[i].z * inv), 0.0f), MAXQ);
        int q3 = (int)fminf(fmaxf(__builtin_rintf(v[i].w * inv), 0.0f), MAXQ);
        lsum += q0 + q1 + q2 + q3;
        w[i] = (uint32_t)q0 | ((uint32_t)q1 << 8) | ((uint32_t)q2 << 16) | ((uint32_t)q3 << 24);
    }
    uint4 packed;
    packed.x = w[0]; packed.y = w[1]; packed.z = w[2]; packed.w = w[3];
    reinterpret_cast<uint4*>(xq + (size_t)row * IN_CH)[tid] = packed;

#pragma unroll
    for (int off = 16; off > 0; off >>= 1) {
        lsum += __shfl_xor(lsum, off, 32);
    }
    if (lane == 0) redi[wave] = lsum;
    __syncthreads();
    if (tid == 0) {
        int s = 0;
#pragma unroll
        for (int i = 0; i < 8; ++i) s += redi[i];
        xscale[row] = scale;
        rowsum[row] = (float)s;
    }
}

__device__ __forceinline__ v8i load_a_frag(gu8* p, int imm)
{
    v8i a;
#pragma unroll
    for (int ppair = 0; ppair < 4; ++ppair) {
        v2i t = *reinterpret_cast<gv2i*>(p + imm + ppair * 8 + (ppair & 2) * 8);
        a[2 * ppair]     = t.x;
        a[2 * ppair + 1] = t.y;
    }
    return a;
}

__device__ __forceinline__ v8i load_b_frag(gu8* p, int imm)
{
    v4i lo = *reinterpret_cast<gv4i*>(p + imm);
    v4i hi = *reinterpret_cast<gv4i*>(p + imm + 32);
    v8i b;
    b[0] = lo.x; b[1] = lo.y; b[2] = lo.z; b[3] = lo.w;
    b[4] = hi.x; b[5] = hi.y; b[6] = hi.z; b[7] = hi.w;
    return b;
}

#define WMMA_IU8(A, B, C) \
    __builtin_amdgcn_wmma_i32_16x16x64_iu8(false, (A), true, (B), (C), false, false)

#define ROWS16 (16 * IN_CH)

__global__ __launch_bounds__(256)
void gemm_iu8_kernel(const uint8_t* __restrict__ xq,
                     const int*     __restrict__ wpk,
                     const float*   __restrict__ wscale,
                     const float*   __restrict__ wzp,
                     const float*   __restrict__ bias,
                     const float*   __restrict__ xscale,
                     const float*   __restrict__ rowsum,
                     float*         __restrict__ out)
{
    const int tid   = threadIdx.x;
    const int wave  = tid >> 5;
    const int lane  = tid & 31;
    const int g     = lane >> 4;
    const int half  = lane & 15;

    const int mBase = blockIdx.x * 256 + wave * 32;
    const int nBase = blockIdx.y * 64;

    gu8* pA = (gu8*)(xq + (size_t)(mBase + half) * IN_CH + g * 16);
    gu8* pW = (gu8*)(reinterpret_cast<const uint8_t*>(wpk) +
                     (size_t)(nBase + half) * IN_CH + g * 16);

    v8i acc[8];
#pragma unroll
    for (int t = 0; t < 8; ++t) acc[t] = (v8i){0, 0, 0, 0, 0, 0, 0, 0};

    v8i aA[2], bA[4], aB[2], bB[4];
    aA[0] = load_a_frag(pA, 0);
    aA[1] = load_a_frag(pA, ROWS16);
#pragma unroll
    for (int j = 0; j < 4; ++j) bA[j] = load_b_frag(pW, j * ROWS16);

#pragma unroll 1
    for (int kt = 0; kt < IN_CH / 64 - 2; kt += 2) {
        aB[0] = load_a_frag(pA, 64);
        aB[1] = load_a_frag(pA, ROWS16 + 64);
#pragma unroll
        for (int j = 0; j < 4; ++j) bB[j] = load_b_frag(pW, j * ROWS16 + 64);

#pragma unroll
        for (int j = 0; j < 4; ++j) {
            acc[j]     = WMMA_IU8(aA[0], bA[j], acc[j]);
            acc[4 + j] = WMMA_IU8(aA[1], bA[j], acc[4 + j]);
        }

        pA += 128;
        pW += 128;
        aA[0] = load_a_frag(pA, 0);
        aA[1] = load_a_frag(pA, ROWS16);
#pragma unroll
        for (int j = 0; j < 4; ++j) bA[j] = load_b_frag(pW, j * ROWS16);

#pragma unroll
        for (int j = 0; j < 4; ++j) {
            acc[j]     = WMMA_IU8(aB[0], bB[j], acc[j]);
            acc[4 + j] = WMMA_IU8(aB[1], bB[j], acc[4 + j]);
        }
    }

    aB[0] = load_a_frag(pA, 64);
    aB[1] = load_a_frag(pA, ROWS16 + 64);
#pragma unroll
    for (int j = 0; j < 4; ++j) bB[j] = load_b_frag(pW, j * ROWS16 + 64);
#pragma unroll
    for (int j = 0; j < 4; ++j) {
        acc[j]     = WMMA_IU8(aA[0], bA[j], acc[j]);
        acc[4 + j] = WMMA_IU8(aA[1], bA[j], acc[4 + j]);
    }
#pragma unroll
    for (int j = 0; j < 4; ++j) {
        acc[j]     = WMMA_IU8(aB[0], bB[j], acc[j]);
        acc[4 + j] = WMMA_IU8(aB[1], bB[j], acc[4 + j]);
    }

    float xs[2][8], rs[2][8];
#pragma unroll
    for (int i = 0; i < 2; ++i) {
#pragma unroll
        for (int r = 0; r < 8; ++r) {
            const int mm = mBase + i * 16 + r + g * 8;
            xs[i][r] = xscale[mm];
            rs[i][r] = rowsum[mm];
        }
    }

#pragma unroll
    for (int j = 0; j < 4; ++j) {
        const int n = nBase + j * 16 + half;
        const float ws = wscale[n];
        const float zp = wzp[n];
        const float bi = bias[n];
#pragma unroll
        for (int i = 0; i < 2; ++i) {
#pragma unroll
            for (int r = 0; r < 8; ++r) {
                const int mm = mBase + i * 16 + r + g * 8;
                const float vv =
                    ((float)acc[i * 4 + j][r] - rs[i][r] * zp) * (xs[i][r] * ws) + bi;
                out[(size_t)mm * OUT_CH + n] = vv;
            }
        }
    }
}

extern "C" void kernel_launch(void* const* d_in, const int* in_sizes, int n_in,
                              void* d_out, int out_size, void* d_ws, size_t ws_size,
                              hipStream_t stream)
{
    const float* x      = (const float*)d_in[0];
    const int*   wpk    = (const int*)  d_in[1];
    const float* wscale = (const float*)d_in[2];
    const float* wzp    = (const float*)d_in[3];
    const float* bias   = (const float*)d_in[4];
    float*       out    = (float*)d_out;

    const int nTokens = in_sizes[0] / IN_CH;

    uint8_t* xq     = (uint8_t*)d_ws;
    float*   xscale = (float*)((char*)d_ws + (size_t)nTokens * IN_CH);
    float*   rowsum = xscale + nTokens;

    quantize_kernel<<<nTokens, 256, 0, stream>>>(x, xq, xscale, rowsum);

    dim3 grid(nTokens / 256, OUT_CH / 64);
    gemm_iu8_kernel<<<grid, 256, 0, stream>>>(xq, wpk, wscale, wzp, bias,
                                              xscale, rowsum, out);
}
